// MultiHeadSelfAttention_83880711290925
// MI455X (gfx1250) — matrix operand order repaired and checked by reading, not compiled, not run on hardware
//
#include <hip/hip_runtime.h>

#define BATCH 4
#define SEQ 2048
#define DMODEL 512
#define NHEADS 8
#define DEPTH 64

typedef __bf16 bf16;
typedef __attribute__((ext_vector_type(16))) __bf16 bf16x16;
typedef __attribute__((ext_vector_type(8)))  __bf16 bf16x8;
typedef __attribute__((ext_vector_type(8)))  float  floatx8;
typedef __attribute__((ext_vector_type(4)))  float  floatx4;

__device__ __forceinline__ floatx8 wmma_bf16(bf16x16 a, bf16x16 b, floatx8 c) {
    return __builtin_amdgcn_wmma_f32_16x16x32_bf16(false, a, false, b, (short)0, c, false, false);
}

// ---------------- WMMA fragment loaders (contiguous k = 16*hi + e order, wave32) ---------
// A 16x32 bf16: lane -> M = lane&15 ; VGPR v holds K pair
//   k = (lane>=16 ? 16 : 0) + v*2
// => per lane: 16 contiguous elems at rowbase+hi*16.
__device__ __forceinline__ bf16x16 load_A_bf16(const bf16* __restrict__ src, int ld, int lane) {
    int m = lane & 15, hi = (lane >> 4) & 1;
    const bf16* rp = src + (size_t)m * ld + (hi << 4);
    bf16x8 lo = *(const bf16x8*)rp;           // 16B aligned -> global_load_b128
    bf16x8 up = *(const bf16x8*)(rp + 8);
    bf16x16 a;
#pragma unroll
    for (int e = 0; e < 8; ++e) { a[e] = lo[e]; a[8 + e] = up[e]; }
    return a;
}

// Non-temporal fp32 variant for the streaming attn matrix (fp32 is mandated:
// attn is a tuple output), converted with v_cvt_pk_bf16_f32 on the fly.
__device__ __forceinline__ bf16x16 load_A_f32_nt(const float* __restrict__ src, int ld, int lane) {
    int m = lane & 15, hi = (lane >> 4) & 1;
    const float* rp = src + (size_t)m * ld + (hi << 4);
    floatx4 f0 = __builtin_nontemporal_load((const floatx4*)(rp));
    floatx4 f1 = __builtin_nontemporal_load((const floatx4*)(rp + 4));
    floatx4 f2 = __builtin_nontemporal_load((const floatx4*)(rp + 8));
    floatx4 f3 = __builtin_nontemporal_load((const floatx4*)(rp + 12));
    bf16x16 a;
#pragma unroll
    for (int e = 0; e < 4; ++e) {
        a[e]      = (bf16)f0[e];
        a[4 + e]  = (bf16)f1[e];
        a[8 + e]  = (bf16)f2[e];
        a[12 + e] = (bf16)f3[e];
    }
    return a;
}

// B 32x16 bf16 from TRANSPOSED source: element (k,n) = src[n*ld + k]
// lanes 0-15: n=lane, K=0..15 ; lanes 16-31: K=16..31 -> 32 contiguous bytes/lane
__device__ __forceinline__ bf16x16 load_Bt_bf16(const bf16* __restrict__ src, int ld, int lane) {
    int n = lane & 15, hi = (lane >> 4) & 1;
    return *(const bf16x16*)(src + (size_t)n * ld + (hi << 4));
}

// ---------------- Kernel 0a: convert x to bf16 (once) ------------------------
// x is re-read 24x by the QKV GEMM; converting once removes all v_cvt + waits
// from that hot loop and halves its A-side bytes. 4.19M elems, 8/thread.
__global__ void __launch_bounds__(256) convert_x_kernel(
    const float* __restrict__ x, bf16* __restrict__ xb) {
    size_t i = ((size_t)blockIdx.x * 256 + threadIdx.x) * 8;
    floatx4 f0 = *(const floatx4*)(x + i);
    floatx4 f1 = *(const floatx4*)(x + i + 4);
    bf16x8 o;
#pragma unroll
    for (int e = 0; e < 4; ++e) { o[e] = (bf16)f0[e]; o[4 + e] = (bf16)f1[e]; }
    *(bf16x8*)(xb + i) = o;
}

// ---------------- Kernel 0b: pack weights into B-fragment order --------------
// wpack[m][ct(32)][kc(16)][lane(32)][e(16)] bf16 ; one thread per (m,ct,kc,lane)
#define FRAGS_PER_MAT 512   // 32 ct * 16 kc
__global__ void __launch_bounds__(256) pack_w_kernel(
    const float* __restrict__ wq, const float* __restrict__ wk,
    const float* __restrict__ wv, const float* __restrict__ wp,
    bf16* __restrict__ wpack) {
    int t = blockIdx.x * 256 + threadIdx.x;   // 65536 threads
    int lane = t & 31, kc = (t >> 5) & 15, ct = (t >> 9) & 31, m = t >> 14;
    const float* W = (m == 0) ? wq : (m == 1) ? wk : (m == 2) ? wv : wp;
    const float* src = W + (size_t)(kc << 5) * DMODEL + (ct << 4);
    int n = lane & 15, hi = (lane >> 4) & 1;
    bf16x16 b;
#pragma unroll
    for (int v = 0; v < 8; ++v) {
        int k0 = (hi << 4) + (v << 1);
        b[2 * v]     = (bf16)src[(size_t)k0 * DMODEL + n];
        b[2 * v + 1] = (bf16)src[(size_t)(k0 + 1) * DMODEL + n];
    }
    *(bf16x16*)(wpack + (((size_t)m * FRAGS_PER_MAT + (ct << 4) + kc) * 32 + lane) * 16) = b;
}

// ---------------- Kernel 1: fused QKV projection -----------------------------
// one wave -> 16 rows x 64 cols (one head); V stored transposed [bh][64][S]
// waves = 3 * 512 * 8 = 12288 -> 1536 blocks
__global__ void __launch_bounds__(256) qkv_proj_kernel(
    const bf16* __restrict__ xb,
    const float* __restrict__ bq, const float* __restrict__ bk,
    const float* __restrict__ bv, const bf16* __restrict__ wpack,
    bf16* __restrict__ qb, bf16* __restrict__ kb, bf16* __restrict__ vbT) {
    int lane   = threadIdx.x & 31;
    int waveId = (blockIdx.x << 3) + (threadIdx.x >> 5);
    int cs = waveId & 7;             // head (64-col strip)
    int rt = (waveId >> 3) & 511;    // 16-row tile
    int m  = waveId >> 12;           // 0=q 1=k 2=v

    const float* Bs = (m == 0) ? bq : (m == 1) ? bk : bv;
    const bf16*  Wp = wpack + (size_t)m * FRAGS_PER_MAT * 512;
    int row0 = rt << 4;

    floatx8 acc[4] = {};
#pragma unroll 2
    for (int kc = 0; kc < 16; ++kc) {
        bf16x16 a = load_A_bf16(xb + (size_t)row0 * DMODEL + (kc << 5), DMODEL, lane);
#pragma unroll
        for (int j = 0; j < 4; ++j) {
            int ct = (cs << 2) + j;
            bf16x16 b = *(const bf16x16*)(Wp + (((size_t)(ct << 4) + kc) * 32 + lane) * 16);
            acc[j] = wmma_bf16(a, b, acc[j]);
        }
    }

    int n = lane & 15, hi = (lane >> 4) & 1;
#pragma unroll
    for (int j = 0; j < 4; ++j) {
        int dp   = (j << 4) + n;             // 0..63 within head
        float bias = Bs[(cs << 6) + dp];
#pragma unroll
        for (int v = 0; v < 8; ++v) {
            int row = row0 + v + (hi << 3);  // token index
            int bI  = row >> 11;
            int s   = row & (SEQ - 1);
            size_t bh = (size_t)bI * NHEADS + cs;
            bf16 val = (bf16)(acc[j][v] + bias);
            if (m == 2) vbT[(bh * DEPTH + dp) * SEQ + s] = val;
            else if (m == 0) qb[(bh * SEQ + s) * DEPTH + dp] = val;
            else             kb[(bh * SEQ + s) * DEPTH + dp] = val;
        }
    }
}

// ---------------- Kernel 2: scores = Q K^T / 8 (NT store) --------------------
// one wave -> 16 queries x 64 keys ; waves = 32*128*32 = 131072 -> 16384 blocks
__global__ void __launch_bounds__(256) scores_kernel(
    const bf16* __restrict__ qb, const bf16* __restrict__ kb,
    float* __restrict__ attn) {
    int lane   = threadIdx.x & 31;
    int waveId = (blockIdx.x << 3) + (threadIdx.x >> 5);
    int ks = waveId & 31;             // 64-key strip
    int qt = (waveId >> 5) & 127;
    int bh = waveId >> 12;

    const bf16* Q = qb + (size_t)bh * SEQ * DEPTH;
    const bf16* K = kb + (size_t)bh * SEQ * DEPTH;
    int q0 = qt << 4, k0 = ks << 6;

    floatx8 acc[4] = {};
#pragma unroll
    for (int kk = 0; kk < DEPTH; kk += 32) {
        bf16x16 a = load_A_bf16(Q + (size_t)q0 * DEPTH + kk, DEPTH, lane);
#pragma unroll
        for (int j = 0; j < 4; ++j) {
            bf16x16 b = load_Bt_bf16(K + (size_t)(k0 + (j << 4)) * DEPTH + kk, DEPTH, lane);
            acc[j] = wmma_bf16(a, b, acc[j]);
        }
    }

    float* dst = attn + (size_t)bh * SEQ * SEQ;
    int n = lane & 15, hi = (lane >> 4) & 1;
#pragma unroll
    for (int j = 0; j < 4; ++j)
#pragma unroll
        for (int v = 0; v < 8; ++v) {
            int row = q0 + v + (hi << 3);
            __builtin_nontemporal_store(acc[j][v] * 0.125f,
                                        dst + (size_t)row * SEQ + k0 + (j << 4) + n);
        }
}

// ---------------- Kernel 3: row softmax (in place, NT, vectorized) -----------
__device__ __forceinline__ float waveMax(float v) {
#pragma unroll
    for (int off = 16; off > 0; off >>= 1) v = fmaxf(v, __shfl_xor(v, off, 32));
    return v;
}
__device__ __forceinline__ float waveSum(float v) {
#pragma unroll
    for (int off = 16; off > 0; off >>= 1) v += __shfl_xor(v, off, 32);
    return v;
}

__global__ void __launch_bounds__(256) softmax_kernel(float* __restrict__ attn) {
    __shared__ float redm[8];
    __shared__ float reds[8];
    float* p = attn + (size_t)blockIdx.x * SEQ + (size_t)threadIdx.x * 8;
    int lane = threadIdx.x & 31, wave = threadIdx.x >> 5;

    floatx4 v0 = __builtin_nontemporal_load((const floatx4*)p);
    floatx4 v1 = __builtin_nontemporal_load((const floatx4*)(p + 4));

    float mx = -__builtin_inff();
#pragma unroll
    for (int i = 0; i < 4; ++i) mx = fmaxf(mx, fmaxf(v0[i], v1[i]));
    mx = waveMax(mx);
    if (lane == 0) redm[wave] = mx;
    __syncthreads();
    mx = redm[0];
#pragma unroll
    for (int i = 1; i < 8; ++i) mx = fmaxf(mx, redm[i]);

    float sum = 0.f;
#pragma unroll
    for (int i = 0; i < 4; ++i) {
        v0[i] = __expf(v0[i] - mx); sum += v0[i];
        v1[i] = __expf(v1[i] - mx); sum += v1[i];
    }
    sum = waveSum(sum);
    if (lane == 0) reds[wave] = sum;
    __syncthreads();
    sum = reds[0];
#pragma unroll
    for (int i = 1; i < 8; ++i) sum += reds[i];

    float inv = __frcp_rn(sum);
#pragma unroll
    for (int i = 0; i < 4; ++i) { v0[i] *= inv; v1[i] *= inv; }
    __builtin_nontemporal_store(v0, (floatx4*)p);
    __builtin_nontemporal_store(v1, (floatx4*)(p + 4));
}

// ---------------- Kernel 4: context = attn @ V (NT attn reads) ---------------
// one wave -> 16 queries x 64 depth (whole head) ; waves = 32*128 = 4096
__global__ void __launch_bounds__(256) av_kernel(
    const float* __restrict__ attn, const bf16* __restrict__ vbT,
    bf16* __restrict__ ctx) {
    int lane   = threadIdx.x & 31;
    int waveId = (blockIdx.x << 3) + (threadIdx.x >> 5);
    int qt = waveId & 127;
    int bh = waveId >> 7;

    const float* A  = attn + (size_t)bh * SEQ * SEQ + (size_t)(qt << 4) * SEQ;
    const bf16*  Vt = vbT + (size_t)bh * DEPTH * SEQ;

    floatx8 acc[4] = {};
    for (int kk = 0; kk < SEQ; kk += 32) {
        bf16x16 a = load_A_f32_nt(A + kk, SEQ, lane);
#pragma unroll
        for (int dt = 0; dt < 4; ++dt) {
            bf16x16 b = load_Bt_bf16(Vt + (size_t)(dt << 4) * SEQ + kk, SEQ, lane);
            acc[dt] = wmma_bf16(a, b, acc[dt]);
        }
    }

    // ctx layout: [B*S, 512] row-major, col = h*64 + dt*16 + n
    int b  = bh >> 3, h = bh & 7;
    int n  = lane & 15, hi = (lane >> 4) & 1;
#pragma unroll
    for (int dt = 0; dt < 4; ++dt) {
        int col = (h << 6) + (dt << 4) + n;
#pragma unroll
        for (int v = 0; v < 8; ++v) {
            int s = (qt << 4) + v + (hi << 3);
            ctx[((size_t)b * SEQ + s) * DMODEL + col] = (bf16)acc[dt][v];
        }
    }
}

// ---------------- Kernel 5: out = ctx @ proj_w + proj_b ----------------------
// one wave -> 16 rows x 64 cols ; waves = 512*8 = 4096 -> 512 blocks
__global__ void __launch_bounds__(256) proj_kernel(
    const bf16* __restrict__ ctx, const bf16* __restrict__ wpack,
    const float* __restrict__ bp, float* __restrict__ out) {
    int lane   = threadIdx.x & 31;
    int waveId = (blockIdx.x << 3) + (threadIdx.x >> 5);
    int cs = waveId & 7;
    int rt = waveId >> 3;
    int row0 = rt << 4;

    const bf16* Wp = wpack + (size_t)3 * FRAGS_PER_MAT * 512;  // proj matrix
    floatx8 acc[4] = {};
#pragma unroll 2
    for (int kc = 0; kc < 16; ++kc) {
        bf16x16 a = load_A_bf16(ctx + (size_t)row0 * DMODEL + (kc << 5), DMODEL, lane);
#pragma unroll
        for (int j = 0; j < 4; ++j) {
            int ct = (cs << 2) + j;
            bf16x16 b = *(const bf16x16*)(Wp + (((size_t)(ct << 4) + kc) * 32 + lane) * 16);
            acc[j] = wmma_bf16(a, b, acc[j]);
        }
    }

    int n = lane & 15, hi = (lane >> 4) & 1;
#pragma unroll
    for (int j = 0; j < 4; ++j) {
        int col = (cs << 6) + (j << 4) + n;
        float bias = bp[col];
#pragma unroll
        for (int v = 0; v < 8; ++v) {
            int row = row0 + v + (hi << 3);
            out[(size_t)row * DMODEL + col] = acc[j][v] + bias;
        }
    }
}

// ---------------- Host side --------------------------------------------------
extern "C" void kernel_launch(void* const* d_in, const int* in_sizes, int n_in,
                              void* d_out, int out_size, void* d_ws, size_t ws_size,
                              hipStream_t stream) {
    const float* x  = (const float*)d_in[0];
    const float* wq = (const float*)d_in[1];
    const float* bq = (const float*)d_in[2];
    const float* wk = (const float*)d_in[3];
    const float* bk = (const float*)d_in[4];
    const float* wv = (const float*)d_in[5];
    const float* bv = (const float*)d_in[6];
    const float* wp = (const float*)d_in[7];
    const float* bp = (const float*)d_in[8];

    float* out  = (float*)d_out;
    float* attn = out + (size_t)BATCH * SEQ * DMODEL;   // tuple output #2

    const size_t QKV_ELEMS = (size_t)BATCH * NHEADS * SEQ * DEPTH; // 4.19M
    bf16* qb    = (bf16*)d_ws;
    bf16* kb    = qb + QKV_ELEMS;
    bf16* vbT   = kb + QKV_ELEMS;                 // [bh][64][S] transposed
    bf16* ctx   = vbT + QKV_ELEMS;                // [B*S, 512]
    bf16* xb    = ctx + QKV_ELEMS;                // [B*S, 512] bf16 copy of x
    bf16* wpack = xb + QKV_ELEMS;                 // 4 * 512 frags * 512 elems = 2 MB
    // total ws use: 5 * 8 MB + 2 MB = 42 MB

    convert_x_kernel<<<2048,  256, 0, stream>>>(x, xb);
    pack_w_kernel   <<<256,   256, 0, stream>>>(wq, wk, wv, wp, wpack);
    qkv_proj_kernel <<<1536,  256, 0, stream>>>(xb, bq, bk, bv, wpack, qb, kb, vbT);
    scores_kernel   <<<16384, 256, 0, stream>>>(qb, kb, attn);
    softmax_kernel  <<<65536, 256, 0, stream>>>(attn);
    av_kernel       <<<512,   256, 0, stream>>>(attn, vbT, ctx);
    proj_kernel     <<<512,   256, 0, stream>>>(ctx, wpack, bp, out);
}
